// TransformerEncoderBlock_35261681500574
// MI455X (gfx1250) — matrix operand order repaired and checked by reading, not compiled, not run on hardware
//
#include <hip/hip_runtime.h>
#include <hip/hip_bf16.h>

// ---------------------------------------------------------------------------
// Types for gfx1250 WMMA
// ---------------------------------------------------------------------------
typedef __attribute__((ext_vector_type(16))) __bf16  v16bf;
typedef __attribute__((ext_vector_type(8)))  float   v8f;
typedef __attribute__((ext_vector_type(8)))  unsigned int v8u;

union FragU { v8u u; v16bf h; };

__device__ inline v8f v8f_zero() {
  v8f z;
  for (int i = 0; i < 8; ++i) z[i] = 0.f;
  return z;
}

__device__ inline unsigned int pack2bf(float lo, float hi) {
  union { __bf16 h[2]; unsigned int u; } p;
  p.h[0] = (__bf16)lo; p.h[1] = (__bf16)hi; return p.u;
}

// A-fragment (16x32 bf16) K index for packed pair v (0..7), lane-half
__device__ inline int a_kidx(int v, int half) {
  return 2 * (v & 3) + ((v & 4) << 2) + (half << 3);
}
// B-fragment (32x16 bf16) K index for packed pair v (0..7), lane-half
__device__ inline int b_kidx(int v, int half) {
  return 2 * (v & 3) + ((v & 4) << 2) + (half << 3);
}

// ---------------------------------------------------------------------------
// DPP-based 16-lane butterfly reductions (VALU only, no DS pipe).
// quad_perm [1,0,3,2] = xor1, quad_perm [2,3,0,1] = xor2,
// row_half_mirror crosses the 4-boundary, row_mirror crosses the 8-boundary.
// After each step values within a subgroup are equal, so any
// boundary-crossing permute completes the reduction.
// ---------------------------------------------------------------------------
template <int CTRL>
__device__ inline float dpp_mov_f32(float x) {
  int xi = __builtin_bit_cast(int, x);
  int r = __builtin_amdgcn_update_dpp(0, xi, CTRL, 0xF, 0xF, true);
  return __builtin_bit_cast(float, r);
}
__device__ inline float red_max16(float x) {
  x = fmaxf(x, dpp_mov_f32<0xB1>(x));   // quad_perm(1,0,3,2)
  x = fmaxf(x, dpp_mov_f32<0x4E>(x));   // quad_perm(2,3,0,1)
  x = fmaxf(x, dpp_mov_f32<0x141>(x));  // row_half_mirror
  x = fmaxf(x, dpp_mov_f32<0x140>(x));  // row_mirror
  return x;
}
__device__ inline float red_sum16(float x) {
  x += dpp_mov_f32<0xB1>(x);
  x += dpp_mov_f32<0x4E>(x);
  x += dpp_mov_f32<0x141>(x);
  x += dpp_mov_f32<0x140>(x);
  return x;
}

// ---------------------------------------------------------------------------
// fp32 -> bf16 elementwise (weights)
// ---------------------------------------------------------------------------
__global__ void cvt_bf16_kernel(const float* __restrict__ in,
                                __bf16* __restrict__ out, int n) {
  int i = blockIdx.x * 256 + threadIdx.x;
  if (i < n) out[i] = (__bf16)in[i];
}

// ---------------------------------------------------------------------------
// LayerNorm (row = 768) -> bf16
// ---------------------------------------------------------------------------
__global__ __launch_bounds__(256) void ln_bf16_kernel(
    const float* __restrict__ x, const float* __restrict__ g,
    const float* __restrict__ b, __bf16* __restrict__ out, int C) {
  __shared__ float red[8];
  const int row = blockIdx.x;
  const float* xr = x + (size_t)row * C;
  float s = 0.f;
  for (int c = threadIdx.x; c < C; c += 256) s += xr[c];
  #pragma unroll
  for (int o = 16; o > 0; o >>= 1) s += __shfl_xor(s, o, 32);
  if ((threadIdx.x & 31) == 0) red[threadIdx.x >> 5] = s;
  __syncthreads();
  float mu = 0.f;
  #pragma unroll
  for (int i = 0; i < 8; ++i) mu += red[i];
  mu /= (float)C;
  __syncthreads();
  float vs = 0.f;
  for (int c = threadIdx.x; c < C; c += 256) { float d = xr[c] - mu; vs += d * d; }
  #pragma unroll
  for (int o = 16; o > 0; o >>= 1) vs += __shfl_xor(vs, o, 32);
  if ((threadIdx.x & 31) == 0) red[threadIdx.x >> 5] = vs;
  __syncthreads();
  float var = 0.f;
  #pragma unroll
  for (int i = 0; i < 8; ++i) var += red[i];
  var /= (float)C;
  const float rstd = rsqrtf(var + 1e-5f);
  for (int c = threadIdx.x; c < C; c += 256)
    out[(size_t)row * C + c] = (__bf16)((xr[c] - mu) * rstd * g[c] + b[c]);
}

// ---------------------------------------------------------------------------
// Generic bf16 WMMA GEMM:  out[M,N] = A[M,K] * W[N,K]^T (+bias)(+gelu)(+resid)
// Block tile 128x128, 8 waves (4x2), wave tile 32x64, K step 32.
// ---------------------------------------------------------------------------
template <bool HAS_BIAS, bool DO_GELU, bool HAS_RESID, bool OUT_BF16>
__global__ __launch_bounds__(256) void gemm_bf16_kernel(
    const __bf16* __restrict__ A, const __bf16* __restrict__ W,
    const float* __restrict__ bias, const float* __restrict__ resid,
    float* __restrict__ outF, __bf16* __restrict__ outH,
    int M, int N, int K) {
  __shared__ __align__(16) __bf16 lds_a[128][40];
  __shared__ __align__(16) __bf16 lds_b[128][40];
  const int t = threadIdx.x;
  const int lane = t & 31, wave = t >> 5;
  const int wm = wave & 3, wn = wave >> 2;
  const int m0 = blockIdx.y * 128, n0 = blockIdx.x * 128;
  const int mlane = lane & 15, half = lane >> 4;

  v8f acc[2][4];
  #pragma unroll
  for (int i = 0; i < 2; ++i)
    #pragma unroll
    for (int j = 0; j < 4; ++j) acc[i][j] = v8f_zero();

  const int lr = t >> 1;
  const int lc = (t & 1) * 16;

  for (int kk = 0; kk < K; kk += 32) {
    __syncthreads();
    {
      const __bf16* sa = A + (size_t)(m0 + lr) * K + kk + lc;
      *(uint4*)&lds_a[lr][lc]     = *(const uint4*)sa;
      *(uint4*)&lds_a[lr][lc + 8] = *(const uint4*)(sa + 8);
      const __bf16* sb = W + (size_t)(n0 + lr) * K + kk + lc;
      *(uint4*)&lds_b[lr][lc]     = *(const uint4*)sb;
      *(uint4*)&lds_b[lr][lc + 8] = *(const uint4*)(sb + 8);
    }
    __syncthreads();

    FragU bfr[4];
    #pragma unroll
    for (int nt = 0; nt < 4; ++nt) {
      const int nrow = wn * 64 + nt * 16 + mlane;
      #pragma unroll
      for (int v = 0; v < 8; ++v)
        bfr[nt].u[v] = *(const unsigned int*)&lds_b[nrow][b_kidx(v, half)];
    }
    FragU afr[2];
    #pragma unroll
    for (int mt = 0; mt < 2; ++mt) {
      const int mrow = wm * 32 + mt * 16 + mlane;
      #pragma unroll
      for (int v = 0; v < 8; ++v)
        afr[mt].u[v] = *(const unsigned int*)&lds_a[mrow][a_kidx(v, half)];
    }
    #pragma unroll
    for (int mt = 0; mt < 2; ++mt)
      #pragma unroll
      for (int nt = 0; nt < 4; ++nt)
        acc[mt][nt] = __builtin_amdgcn_wmma_f32_16x16x32_bf16(
            false, afr[mt].h, false, bfr[nt].h, (short)0, acc[mt][nt],
            false, false);
  }

  // epilogue: C frag lane mapping: row = r + 8*half, col = lane&15
  #pragma unroll
  for (int mt = 0; mt < 2; ++mt) {
    #pragma unroll
    for (int nt = 0; nt < 4; ++nt) {
      const int col = n0 + wn * 64 + nt * 16 + mlane;
      float bv = HAS_BIAS ? bias[col] : 0.f;
      #pragma unroll
      for (int r = 0; r < 8; ++r) {
        const int row = m0 + wm * 32 + mt * 16 + r + (half << 3);
        float v = acc[mt][nt][r] + bv;
        if (DO_GELU) v = 0.5f * v * (1.f + erff(v * 0.70710678118654752f));
        if (HAS_RESID) v += resid[(size_t)row * N + col];
        if (OUT_BF16) outH[(size_t)row * N + col] = (__bf16)v;
        else          outF[(size_t)row * N + col] = v;
      }
    }
  }
}

// ---------------------------------------------------------------------------
// Transpose V part of qkv -> vT[b,h,d,key]
// ---------------------------------------------------------------------------
__global__ __launch_bounds__(256) void transpose_v_kernel(
    const __bf16* __restrict__ qkv, __bf16* __restrict__ vT) {
  __shared__ __align__(16) __bf16 tile[64][72];
  const int bh = blockIdx.y;
  const int b = bh / 12, h = bh % 12;
  const int ntile = blockIdx.x * 64;
  const int t = threadIdx.x;
  {
    const int nl = t >> 2;
    const int d0 = (t & 3) * 16;
    const __bf16* src =
        qkv + (size_t)(b * 2048 + ntile + nl) * 2304 + 1536 + h * 64 + d0;
    *(uint4*)&tile[nl][d0]     = *(const uint4*)src;
    *(uint4*)&tile[nl][d0 + 8] = *(const uint4*)(src + 8);
  }
  __syncthreads();
  {
    const int dl = t >> 2;
    const int n0 = (t & 3) * 16;
    __bf16* dst = vT + (size_t)(bh * 64 + dl) * 2048 + ntile + n0;
    #pragma unroll
    for (int i = 0; i < 8; ++i) {
      unsigned int p = pack2bf((float)tile[n0 + 2 * i][dl],
                               (float)tile[n0 + 2 * i + 1][dl]);
      *(unsigned int*)(dst + 2 * i) = p;
    }
  }
}

// ---------------------------------------------------------------------------
// Flash-style attention: one wave = 16 query rows of one (b,h).
// S tile 16x32 per iter (4 WMMA), online softmax (DPP reductions), P*V (4 WMMA).
// ---------------------------------------------------------------------------
__global__ __launch_bounds__(128) void attention_kernel(
    const __bf16* __restrict__ qkv, const __bf16* __restrict__ vT,
    __bf16* __restrict__ out) {
  __shared__ __align__(4) __bf16 pstage[4][16][34];
  const int t = threadIdx.x;
  const int lane = t & 31, wave = t >> 5;
  const int mlane = lane & 15, half = lane >> 4;
  const int bh = blockIdx.y, b = bh / 12, h = bh % 12;
  const int q0 = blockIdx.x * 64 + wave * 16;
  const size_t TS = 2304;
  const __bf16* qbase = qkv + (size_t)(b * 2048) * TS + h * 64;
  const __bf16* kbase = qkv + (size_t)(b * 2048) * TS + 768 + h * 64;
  const __bf16* vtb   = vT + (size_t)(bh * 64) * 2048;

  FragU aq[2];
  #pragma unroll
  for (int kf = 0; kf < 2; ++kf)
    #pragma unroll
    for (int v = 0; v < 8; ++v) {
      int d = kf * 32 + a_kidx(v, half);
      aq[kf].u[v] =
          *(const unsigned int*)(qbase + (size_t)(q0 + mlane) * TS + d);
    }

  v8f acc[4];
  #pragma unroll
  for (int i = 0; i < 4; ++i) acc[i] = v8f_zero();
  float mrun[8], lrun[8];
  #pragma unroll
  for (int r = 0; r < 8; ++r) { mrun[r] = -1e30f; lrun[r] = 0.f; }

  for (int j = 0; j < 2048; j += 32) {
    v8f s[2];
    #pragma unroll
    for (int c = 0; c < 2; ++c) {
      FragU kb[2];
      #pragma unroll
      for (int kf = 0; kf < 2; ++kf)
        #pragma unroll
        for (int v = 0; v < 8; ++v) {
          int d = kf * 32 + b_kidx(v, half);
          kb[kf].u[v] = *(const unsigned int*)(
              kbase + (size_t)(j + c * 16 + mlane) * TS + d);
        }
      v8f z = v8f_zero();
      z = __builtin_amdgcn_wmma_f32_16x16x32_bf16(
          false, aq[0].h, false, kb[0].h, (short)0, z, false, false);
      z = __builtin_amdgcn_wmma_f32_16x16x32_bf16(
          false, aq[1].h, false, kb[1].h, (short)0, z, false, false);
      s[c] = z;
    }

    float alpha[8];
    #pragma unroll
    for (int r = 0; r < 8; ++r) {
      float a = s[0][r] * 0.125f, bb = s[1][r] * 0.125f;
      float mx = red_max16(fmaxf(a, bb));
      float mnew = fmaxf(mrun[r], mx);
      alpha[r] = __expf(mrun[r] - mnew);
      mrun[r] = mnew;
      float p0 = __expf(a - mnew);
      float p1 = __expf(bb - mnew);
      s[0][r] = p0; s[1][r] = p1;
      float rs = red_sum16(p0 + p1);
      lrun[r] = lrun[r] * alpha[r] + rs;
    }
    #pragma unroll
    for (int nt = 0; nt < 4; ++nt)
      #pragma unroll
      for (int r = 0; r < 8; ++r) acc[nt][r] *= alpha[r];

    // P: C-layout -> LDS -> A-layout bf16 fragment
    #pragma unroll
    for (int c = 0; c < 2; ++c)
      #pragma unroll
      for (int r = 0; r < 8; ++r)
        pstage[wave][r + (half << 3)][c * 16 + mlane] = (__bf16)s[c][r];
    FragU ap;
    #pragma unroll
    for (int v = 0; v < 8; ++v)
      ap.u[v] = *(const unsigned int*)&pstage[wave][mlane][a_kidx(v, half)];

    #pragma unroll
    for (int nt = 0; nt < 4; ++nt) {
      FragU vb;
      #pragma unroll
      for (int v = 0; v < 8; ++v) {
        int key = j + b_kidx(v, half);
        vb.u[v] = *(const unsigned int*)(
            vtb + (size_t)(nt * 16 + mlane) * 2048 + key);
      }
      acc[nt] = __builtin_amdgcn_wmma_f32_16x16x32_bf16(
          false, ap.h, false, vb.h, (short)0, acc[nt], false, false);
    }
  }

  #pragma unroll
  for (int nt = 0; nt < 4; ++nt) {
    const int col = h * 64 + nt * 16 + mlane;
    #pragma unroll
    for (int r = 0; r < 8; ++r) {
      const int row = q0 + r + (half << 3);
      out[(size_t)(b * 2048 + row) * 768 + col] =
          (__bf16)(acc[nt][r] / lrun[r]);
    }
  }
}

// ---------------------------------------------------------------------------
// Host-side orchestration
// ---------------------------------------------------------------------------
extern "C" void kernel_launch(void* const* d_in, const int* in_sizes, int n_in,
                              void* d_out, int out_size, void* d_ws,
                              size_t ws_size, hipStream_t stream) {
  (void)in_sizes; (void)n_in; (void)out_size; (void)ws_size;
  const int B = 2, N = 2048, C = 768, H = 12, MH = 3072;
  const int M = B * N;  // 4096 tokens

  const float* x      = (const float*)d_in[0];
  const float* ln1_g  = (const float*)d_in[1];
  const float* ln1_b  = (const float*)d_in[2];
  const float* qkv_w  = (const float*)d_in[3];
  const float* proj_w = (const float*)d_in[4];
  const float* proj_b = (const float*)d_in[5];
  const float* ln2_g  = (const float*)d_in[6];
  const float* ln2_b  = (const float*)d_in[7];
  const float* fc1_w  = (const float*)d_in[8];
  const float* fc1_b  = (const float*)d_in[9];
  const float* fc2_w  = (const float*)d_in[10];
  const float* fc2_b  = (const float*)d_in[11];
  float* out = (float*)d_out;

  // workspace layout
  size_t off = 0;
  auto alloc = [&](size_t bytes) {
    size_t o = off; off = (off + bytes + 255) & ~(size_t)255; return o;
  };
  char* ws = (char*)d_ws;
  __bf16* wq   = (__bf16*)(ws + alloc((size_t)3 * C * C * 2));
  __bf16* wp   = (__bf16*)(ws + alloc((size_t)C * C * 2));
  __bf16* w1   = (__bf16*)(ws + alloc((size_t)MH * C * 2));
  __bf16* w2   = (__bf16*)(ws + alloc((size_t)C * MH * 2));
  __bf16* hbuf = (__bf16*)(ws + alloc((size_t)M * C * 2));
  __bf16* qkv  = (__bf16*)(ws + alloc((size_t)M * 3 * C * 2));
  __bf16* vT   = (__bf16*)(ws + alloc((size_t)B * H * 64 * N * 2));
  __bf16* aout = (__bf16*)(ws + alloc((size_t)M * C * 2));
  float*  x1   = (float*)(ws + alloc((size_t)M * C * 4));
  __bf16* hmid = (__bf16*)(ws + alloc((size_t)M * MH * 2));

  // 1) weights -> bf16
  {
    int n;
    n = 3 * C * C; cvt_bf16_kernel<<<n / 256, 256, 0, stream>>>(qkv_w, wq, n);
    n = C * C;     cvt_bf16_kernel<<<n / 256, 256, 0, stream>>>(proj_w, wp, n);
    n = MH * C;    cvt_bf16_kernel<<<n / 256, 256, 0, stream>>>(fc1_w, w1, n);
    n = C * MH;    cvt_bf16_kernel<<<n / 256, 256, 0, stream>>>(fc2_w, w2, n);
  }
  // 2) LN1
  ln_bf16_kernel<<<M, 256, 0, stream>>>(x, ln1_g, ln1_b, hbuf, C);
  // 3) QKV gemm: [M,768]x[2304,768]^T -> bf16
  gemm_bf16_kernel<false, false, false, true>
      <<<dim3(3 * C / 128, M / 128), 256, 0, stream>>>(
          hbuf, wq, nullptr, nullptr, nullptr, qkv, M, 3 * C, C);
  // 4) transpose V
  transpose_v_kernel<<<dim3(N / 64, B * H), 256, 0, stream>>>(qkv, vT);
  // 5) attention
  attention_kernel<<<dim3(N / 64, B * H), 128, 0, stream>>>(qkv, vT, aout);
  // 6) proj gemm + bias + residual(x) -> fp32 x1
  gemm_bf16_kernel<true, false, true, false>
      <<<dim3(C / 128, M / 128), 256, 0, stream>>>(
          aout, wp, proj_b, x, x1, nullptr, M, C, C);
  // 7) LN2
  ln_bf16_kernel<<<M, 256, 0, stream>>>(x1, ln2_g, ln2_b, hbuf, C);
  // 8) FC1 + bias + exact GELU -> bf16
  gemm_bf16_kernel<true, true, false, true>
      <<<dim3(MH / 128, M / 128), 256, 0, stream>>>(
          hbuf, w1, fc1_b, nullptr, nullptr, hmid, M, MH, C);
  // 9) FC2 + bias + residual(x1) -> fp32 out
  gemm_bf16_kernel<true, false, true, false>
      <<<dim3(C / 128, M / 128), 256, 0, stream>>>(
          hmid, w2, fc2_b, x1, out, nullptr, M, C, MH);
}
